// IdealDPM_old_58626303590989
// MI455X (gfx1250) — compile-verified, matrix operand order checked by reading, not run on hardware
//
#include <hip/hip_runtime.h>
#include <hip/hip_bf16.h>
#include <math.h>

#define B_ 64
#define T_ 1024
#define IN_DIM_ 7
#define EMB_ 64
#define H_ 2
#define DH_ 32
#define THRESH_ 0.2f

typedef __attribute__((ext_vector_type(16))) _Float16 v16h;
typedef __attribute__((ext_vector_type(8)))  float    v8f;

// Load an A/B 16x32 f16 WMMA fragment from a row-major [rows][rowStride] f16 array.
// Lane L (wave32): row = l16 = L&15 ; group g = L>>4.
// halves j<8  -> col = colBase + g*8 + j
// halves j>=8 -> col = colBase + 16 + g*8 + (j-8)
// Both 8-half groups are contiguous -> two 16B loads per lane.
__device__ __forceinline__ v16h load_ab_frag(const _Float16* rowBase, int rowStride, int colBase) {
  const int lane = threadIdx.x & 31;
  const int l16  = lane & 15;
  const int g    = lane >> 4;
  const _Float16* r = rowBase + (size_t)l16 * rowStride + colBase + g * 8;
  union { v16h h; uint4 u[2]; } f;
  f.u[0] = *(const uint4*)(r);
  f.u[1] = *(const uint4*)(r + 16);
  return f.h;
}

// Per-lane prefetch of the same addresses a later load_ab_frag will touch.
// Lowers to global_prefetch_b8 (one byte per lane pulls the whole cacheline).
__device__ __forceinline__ void prefetch_ab_frag(const _Float16* rowBase, int rowStride, int colBase) {
  const int lane = threadIdx.x & 31;
  const int l16  = lane & 15;
  const int g    = lane >> 4;
  const _Float16* r = rowBase + (size_t)l16 * rowStride + colBase + g * 8;
  __builtin_prefetch(r, 0, 1);
}

// ---------------- Kernel 1: h = xW_in^T + b ; qkv = hW^T + b ; emit f16 Q,K (row major
// [b,h,t,dh]) and V transposed ([b,h,dh,t]); 1/sqrt(dh) folded into Q. --------------
__global__ void qkv_kernel(const float* __restrict__ x, const float* __restrict__ w_in,
                           const float* __restrict__ b_in, const float* __restrict__ ipw,
                           const float* __restrict__ ipb,
                           _Float16* __restrict__ Q, _Float16* __restrict__ K,
                           _Float16* __restrict__ Vt) {
  __shared__ float s_win[EMB_ * IN_DIM_];
  __shared__ float s_bin[EMB_];
  __shared__ float s_ipw[3 * EMB_ * EMB_];
  __shared__ float s_ipb[3 * EMB_];
  for (int i = threadIdx.x; i < EMB_ * IN_DIM_; i += blockDim.x) s_win[i] = w_in[i];
  for (int i = threadIdx.x; i < EMB_;           i += blockDim.x) s_bin[i] = b_in[i];
  for (int i = threadIdx.x; i < 3 * EMB_ * EMB_; i += blockDim.x) s_ipw[i] = ipw[i];
  for (int i = threadIdx.x; i < 3 * EMB_;       i += blockDim.x) s_ipb[i] = ipb[i];
  __syncthreads();

  const int tok = blockIdx.x * blockDim.x + threadIdx.x;   // B_*T_ threads total
  const int b = tok / T_, t = tok % T_;

  float xv[IN_DIM_];
  const float* xp = x + (size_t)tok * IN_DIM_;
  for (int i = 0; i < IN_DIM_; ++i) xv[i] = xp[i];

  float h[EMB_];
  for (int e = 0; e < EMB_; ++e) {
    float a = s_bin[e];
    for (int i = 0; i < IN_DIM_; ++i) a += xv[i] * s_win[e * IN_DIM_ + i];
    h[e] = a;
  }

  const float qscale = 0.17677669529663687f; // 1/sqrt(32)
  for (int e = 0; e < 3 * EMB_; ++e) {
    float a = s_ipb[e];
    const float* w = &s_ipw[e * EMB_];
    for (int j = 0; j < EMB_; ++j) a += h[j] * w[j];
    if (e < EMB_) {
      int hd = e / DH_, d = e % DH_;
      Q[(((size_t)(b * H_ + hd)) * T_ + t) * DH_ + d] = (_Float16)(a * qscale);
    } else if (e < 2 * EMB_) {
      int ee = e - EMB_; int hd = ee / DH_, d = ee % DH_;
      K[(((size_t)(b * H_ + hd)) * T_ + t) * DH_ + d] = (_Float16)a;
    } else {
      int ee = e - 2 * EMB_; int hd = ee / DH_, d = ee % DH_;
      Vt[(((size_t)(b * H_ + hd)) * DH_ + d) * T_ + t] = (_Float16)a;
    }
  }
}

// ---------------- Kernel 2: flash attention, one 16-query tile per wave. ----------
// Computes S^T = K(16x32) x Q^T(32x16) via v_wmma_f32_16x16x32_f16 (lane = query col,
// VGPRs = k rows), online softmax with a single shfl_xor(16) partner reduction, then
// ctx^T += V^T(16x32) x P^T(32x16). The P^T D-fragments map directly onto the B
// operand half layout -> no cross-lane shuffles for P. Unroll-2 + prefetch overlap
// the next tile's loads with the current tile's softmax VALU (WMMA co-executes).
__global__ void attn_kernel(const _Float16* __restrict__ Q, const _Float16* __restrict__ K,
                            const _Float16* __restrict__ Vt, float* __restrict__ ctx) {
  const int wave = threadIdx.x >> 5;
  const int tile = blockIdx.x * (blockDim.x >> 5) + wave;  // 0..B_*H_*(T_/16)-1
  const int qt = tile & 63;
  const int hd = (tile >> 6) & (H_ - 1);
  const int b  = tile >> 7;
  const int lane = threadIdx.x & 31;
  const int l16 = lane & 15, g = lane >> 4;

  const size_t bh = (size_t)(b * H_ + hd);
  const _Float16* Qb = Q  + bh * T_ * DH_;
  const _Float16* Kb = K  + bh * T_ * DH_;
  const _Float16* Vb = Vt + bh * DH_ * T_;
  const int qb = qt * 16;

  const v16h bQ = load_ab_frag(Qb + (size_t)qb * DH_, DH_, 0);  // B operand: Q^T
  v8f o0 = {}, o1 = {};
  float m = -INFINITY, l = 0.f;

#pragma unroll 2
  for (int kb = 0; kb < T_; kb += 32) {
    // Pull next 32-k tile of K and V^T toward the WGP while this tile computes.
    if (kb + 32 < T_) {
      prefetch_ab_frag(Kb + (size_t)(kb + 32) * DH_, DH_, 0);
      prefetch_ab_frag(Kb + (size_t)(kb + 48) * DH_, DH_, 0);
      prefetch_ab_frag(Vb,                  T_, kb + 32);
      prefetch_ab_frag(Vb + (size_t)16 * T_, T_, kb + 32);
    }

    v16h aK0 = load_ab_frag(Kb + (size_t)kb * DH_, DH_, 0);
    v16h aK1 = load_ab_frag(Kb + (size_t)(kb + 16) * DH_, DH_, 0);
    v16h aV0 = load_ab_frag(Vb,                  T_, kb);   // V^T rows dh 0..15
    v16h aV1 = load_ab_frag(Vb + (size_t)16 * T_, T_, kb);  // V^T rows dh 16..31

    v8f z = {};
    v8f s0 = __builtin_amdgcn_wmma_f32_16x16x32_f16(false, aK0, false, bQ, (short)0, z, false, false);
    v8f s1 = __builtin_amdgcn_wmma_f32_16x16x32_f16(false, aK1, false, bQ, (short)0, z, false, false);

    float bm = -INFINITY;
    for (int j = 0; j < 8; ++j) { bm = fmaxf(bm, s0[j]); bm = fmaxf(bm, s1[j]); }
    bm = fmaxf(bm, __shfl_xor(bm, 16, 32));        // combine partner lane (other k half)
    float m_new = fmaxf(m, bm);
    float scale = __expf(m - m_new);

    float ls = 0.f;
    v16h pB;                                        // P^T as B operand (K=32 over k)
    for (int j = 0; j < 8; ++j) {
      float e0 = __expf(s0[j] - m_new);
      float e1 = __expf(s1[j] - m_new);
      ls += e0 + e1;
      pB[j]     = (_Float16)e0;   // k = kb + g*8 + j
      pB[j + 8] = (_Float16)e1;   // k = kb + 16 + g*8 + j
    }
    l = l * scale + ls;
    for (int j = 0; j < 8; ++j) { o0[j] *= scale; o1[j] *= scale; }

    o0 = __builtin_amdgcn_wmma_f32_16x16x32_f16(false, aV0, false, pB, (short)0, o0, false, false);
    o1 = __builtin_amdgcn_wmma_f32_16x16x32_f16(false, aV1, false, pB, (short)0, o1, false, false);
    m = m_new;
  }

  l += __shfl_xor(l, 16, 32);
  const float inv = 1.f / l;
  // ctx^T frag: lane -> q col = qb+l16; VGPR r -> dh = (tile)*16 + r + 8*g
  float* cp = ctx + ((size_t)(b * T_ + qb + l16)) * EMB_ + hd * DH_ + g * 8;
  for (int r = 0; r < 8; ++r) cp[r]      = o0[r] * inv;
  for (int r = 0; r < 8; ++r) cp[16 + r] = o1[r] * inv;
}

// ---------------- Kernel 3: attn_out = ctx W_out^T + b ; boundary predictor. ------
__global__ void proj_bp_kernel(const float* __restrict__ ctx, const float* __restrict__ out_w,
                               const float* __restrict__ out_b, const float* __restrict__ bpw1,
                               const float* __restrict__ bpb1, const float* __restrict__ bpw2,
                               const float* __restrict__ bpb2,
                               float* __restrict__ ao, int* __restrict__ starts) {
  __shared__ float s_ow[EMB_ * EMB_];
  __shared__ float s_ob[EMB_];
  __shared__ float s_w1[32 * EMB_];
  __shared__ float s_b1[32];
  __shared__ float s_w2[32];
  __shared__ float s_b2;
  for (int i = threadIdx.x; i < EMB_ * EMB_; i += blockDim.x) s_ow[i] = out_w[i];
  for (int i = threadIdx.x; i < EMB_;        i += blockDim.x) s_ob[i] = out_b[i];
  for (int i = threadIdx.x; i < 32 * EMB_;   i += blockDim.x) s_w1[i] = bpw1[i];
  for (int i = threadIdx.x; i < 32;          i += blockDim.x) { s_b1[i] = bpb1[i]; s_w2[i] = bpw2[i]; }
  if (threadIdx.x == 0) s_b2 = bpb2[0];
  __syncthreads();

  const int tok = blockIdx.x * blockDim.x + threadIdx.x;
  const int t = tok % T_;

  float c[EMB_];
  const float* cp = ctx + (size_t)tok * EMB_;
  for (int j = 0; j < EMB_; ++j) c[j] = cp[j];

  float a[EMB_];
  for (int e = 0; e < EMB_; ++e) {
    float s = s_ob[e];
    const float* w = &s_ow[e * EMB_];
    for (int j = 0; j < EMB_; ++j) s += c[j] * w[j];
    a[e] = s;
  }
  float* ap = ao + (size_t)tok * EMB_;
  for (int e = 0; e < EMB_; ++e) ap[e] = a[e];

  float s = s_b2;
  for (int o = 0; o < 32; ++o) {
    float r = s_b1[o];
    const float* w = &s_w1[o * EMB_];
    for (int j = 0; j < EMB_; ++j) r += a[j] * w[j];
    r = fmaxf(r, 0.f);
    s += r * s_w2[o];
  }
  const float sig = 1.f / (1.f + __expf(-s));
  const int st = (t == 0) ? 1 : ((t >= 1 && t <= T_ - 2 && sig > THRESH_) ? 1 : 0);
  starts[tok] = st;
}

// ---------------- Kernel 4: segment mean pooling (deterministic, no float atomics).
__global__ void segment_kernel(const float* __restrict__ ao, const int* __restrict__ starts,
                               float* __restrict__ means) {
  __shared__ int s_st[T_];
  __shared__ int s_seg[T_];
  const int b = blockIdx.x;
  for (int i = threadIdx.x; i < T_; i += blockDim.x) s_st[i] = starts[b * T_ + i];
  __syncthreads();
  if (threadIdx.x == 0) {
    int c = 0;
    for (int i = 0; i < T_; ++i) { c += s_st[i]; s_seg[i] = c - 1; }
  }
  __syncthreads();

  float* mb = means + (size_t)b * T_ * EMB_;
  for (int i = threadIdx.x; i < T_ * EMB_; i += blockDim.x) mb[i] = 0.f;
  __syncthreads();

  for (int t = threadIdx.x; t < T_; t += blockDim.x) {
    if (s_st[t]) {
      int end = t + 1;
      while (end < T_ && !s_st[end]) ++end;
      float sum[EMB_];
      for (int e = 0; e < EMB_; ++e) sum[e] = 0.f;
      const float* arow = ao + ((size_t)b * T_ + t) * EMB_;
      for (int r = t; r < end; ++r) {
        for (int e = 0; e < EMB_; ++e) sum[e] += arow[e];
        arow += EMB_;
      }
      const float invc = 1.f / (float)(end - t);
      float* mrow = mb + (size_t)s_seg[t] * EMB_;
      for (int e = 0; e < EMB_; ++e) mrow[e] = sum[e] * invc;
    }
  }
}

// ---------------- Kernel 5: out = means W_proj^T + b. ----------------------------
__global__ void final_proj_kernel(const float* __restrict__ means, const float* __restrict__ pw,
                                  const float* __restrict__ pb, float* __restrict__ out) {
  __shared__ float s_w[EMB_ * EMB_];
  __shared__ float s_b[EMB_];
  for (int i = threadIdx.x; i < EMB_ * EMB_; i += blockDim.x) s_w[i] = pw[i];
  for (int i = threadIdx.x; i < EMB_;        i += blockDim.x) s_b[i] = pb[i];
  __syncthreads();

  const int tok = blockIdx.x * blockDim.x + threadIdx.x;
  float mrow[EMB_];
  const float* mp = means + (size_t)tok * EMB_;
  for (int j = 0; j < EMB_; ++j) mrow[j] = mp[j];

  float* op = out + (size_t)tok * EMB_;
  for (int e = 0; e < EMB_; ++e) {
    float s = s_b[e];
    const float* w = &s_w[e * EMB_];
    for (int j = 0; j < EMB_; ++j) s += mrow[j] * w[j];
    op[e] = s;
  }
}

extern "C" void kernel_launch(void* const* d_in, const int* in_sizes, int n_in,
                              void* d_out, int out_size, void* d_ws, size_t ws_size,
                              hipStream_t stream) {
  const float* x    = (const float*)d_in[0];
  const float* w_in = (const float*)d_in[1];
  const float* b_in = (const float*)d_in[2];
  const float* ipw  = (const float*)d_in[3];
  const float* ipb  = (const float*)d_in[4];
  const float* outw = (const float*)d_in[5];
  const float* outb = (const float*)d_in[6];
  const float* bpw1 = (const float*)d_in[7];
  const float* bpb1 = (const float*)d_in[8];
  const float* bpw2 = (const float*)d_in[9];
  const float* bpb2 = (const float*)d_in[10];
  const float* pw   = (const float*)d_in[11];
  const float* pb   = (const float*)d_in[12];
  float* out = (float*)d_out;

  char* ws = (char*)d_ws;
  size_t off = 0;
  auto wsalloc = [&](size_t bytes) -> void* {
    void* p = ws + off;
    off += (bytes + 255) & ~(size_t)255;
    return p;
  };
  _Float16* Q   = (_Float16*)wsalloc((size_t)B_ * H_ * T_ * DH_ * 2);   // 8 MB
  _Float16* K   = (_Float16*)wsalloc((size_t)B_ * H_ * T_ * DH_ * 2);   // 8 MB
  _Float16* Vt  = (_Float16*)wsalloc((size_t)B_ * H_ * DH_ * T_ * 2);   // 8 MB
  float* ctx    = (float*)wsalloc((size_t)B_ * T_ * EMB_ * 4);          // 16 MB
  float* ao     = (float*)wsalloc((size_t)B_ * T_ * EMB_ * 4);          // 16 MB
  int* starts   = (int*)wsalloc((size_t)B_ * T_ * 4);                   // 256 KB
  float* means  = (float*)wsalloc((size_t)B_ * T_ * EMB_ * 4);          // 16 MB
  (void)ws_size; (void)in_sizes; (void)n_in; (void)out_size;

  const dim3 blk(256);
  qkv_kernel<<<(B_ * T_) / 256, blk, 0, stream>>>(x, w_in, b_in, ipw, ipb, Q, K, Vt);
  attn_kernel<<<(B_ * H_ * (T_ / 16)) / 8, blk, 0, stream>>>(Q, K, Vt, ctx);
  proj_bp_kernel<<<(B_ * T_) / 256, blk, 0, stream>>>(ctx, outw, outb, bpw1, bpb1, bpw2, bpb2, ao, starts);
  segment_kernel<<<B_, blk, 0, stream>>>(ao, starts, means);
  final_proj_kernel<<<(B_ * T_) / 256, blk, 0, stream>>>(means, pw, pb, out);
}
